// RetNetAttentionWrapper_33809982554540
// MI455X (gfx1250) — compile-verified, matrix operand order checked by reading, not run on hardware
//
#include <hip/hip_runtime.h>
#include <math.h>

// RetNet attention for MI455X (gfx1250, wave32, WMMA bf16 16x16x32, f32 accum).
// B=4 T=4096 D=1024 H=8 DK=128 DV=256 CHUNK=64
#define B_   4
#define T_   4096
#define D_   1024
#define H_   8
#define DK_  128
#define DV_  256
#define C_   64
#define NCH_ 64

typedef __bf16 bf16;
typedef __attribute__((ext_vector_type(16))) __bf16 v16bf;
typedef __attribute__((ext_vector_type(8)))  __bf16 v8bf;
typedef __attribute__((ext_vector_type(4)))  __bf16 bf16x4;
typedef __attribute__((ext_vector_type(2)))  __bf16 v2bf;
typedef __attribute__((ext_vector_type(8)))  float  v8f;
typedef int i32x4 __attribute__((vector_size(16)));  // matches async-LDS builtin pointee

__device__ __forceinline__ v8f wmma_bf16(v16bf a, v16bf b, v8f c) {
  // D = A x B + C, 16x16x32 bf16 -> f32
  return __builtin_amdgcn_wmma_f32_16x16x32_bf16(false, a, false, b, (short)0, c,
                                                 false, false);
}

// A-fragment (16x32, bf16) from row-major LDS tile p[ld]: lanes 0-15 hold M=lane,
// K={0..7,16..23}; lanes 16-31 hold M=lane-16, K={8..15,24..31}.
// ld MUST be a multiple of 8 elements and p 16B-aligned so both halves are
// contiguous 16B chunks -> two ds_load_b128 per fragment.
__device__ __forceinline__ v16bf load_frag_a(const bf16* p, int ld, int m0, int k0) {
  const int lane = threadIdx.x & 31;
  const bf16* row = p + (m0 + (lane & 15)) * ld + k0 + ((lane >> 4) << 3);
  const v8bf lo = *(const v8bf*)(row);
  const v8bf hi = *(const v8bf*)(row + 16);
  return __builtin_shufflevector(lo, hi, 0, 1, 2, 3, 4, 5, 6, 7,
                                 8, 9, 10, 11, 12, 13, 14, 15);
}

// ---------------- f32 -> bf16 bulk convert (vectorized x4) ----------------
__global__ __launch_bounds__(256) void cvt_f32_to_bf16(const float* __restrict__ in,
                                                       bf16* __restrict__ out, int n4) {
  int i = blockIdx.x * 256 + threadIdx.x;
  if (i >= n4) return;
  const float4 f = ((const float4*)in)[i];
  bf16x4 o = {(bf16)f.x, (bf16)f.y, (bf16)f.z, (bf16)f.w};
  ((bf16x4*)out)[i] = o;
}

// ---------------- tiled bf16 GEMM: C[M,N] = A[M,K] @ B[K,N] ----------------
// 128x128 tile, BK=64 (2 WMMA k-steps per barrier pair), 256 threads = 8 waves,
// each wave 32x64 (2x4 WMMA tiles, 16 WMMAs per slab).
// B is staged TRANSPOSED (N x K) in LDS so both A and B fragments are b128 loads.
__global__ __launch_bounds__(256) void gemm_bf16_wmma(
    const bf16* __restrict__ A, const bf16* __restrict__ Bm,
    float* __restrict__ Cm, int M, int N, int K) {
  (void)M;
  __shared__ __align__(16) bf16 sA[128 * 64];   // M x K slab (16 KB)
  __shared__ __align__(16) bf16 sBt[128 * 72];  // N x K slab, pad 8 (18 KB)
  const int tid = threadIdx.x;
  const int wave = tid >> 5, wm = wave >> 1, wn = wave & 1;
  const int lane = tid & 31, hi = lane >> 4, cc = lane & 15;
  const int tile_m = blockIdx.y * 128, tile_n = blockIdx.x * 128;
  v8f acc[2][4] = {};
  const int ar = tid >> 2, ac = (tid & 3) << 4;   // A: 4 thr/row, rows ar, ar+64
  const int bk = (tid >> 3) << 1;                 // B: k-pair 0,2,..,62
  const int bn = (tid & 7) << 4;                  // B: 16 n-cols per thread
  for (int k0 = 0; k0 < K; k0 += 64) {
    {  // A slab: two rows x 16 contiguous elems = 4x b128
      const uint4* ga0 = (const uint4*)(A + (size_t)(tile_m + ar) * K + k0 + ac);
      const uint4* ga1 = (const uint4*)(A + (size_t)(tile_m + ar + 64) * K + k0 + ac);
      uint4* la0 = (uint4*)(sA + ar * 64 + ac);
      uint4* la1 = (uint4*)(sA + (ar + 64) * 64 + ac);
      la0[0] = ga0[0]; la0[1] = ga0[1];
      la1[0] = ga1[0]; la1[1] = ga1[1];
    }
    {  // B transpose-in-LDS: two K-rows coalesced, pack (k,k+1) -> b32 stores
      const bf16* g0 = Bm + (size_t)(k0 + bk) * N + tile_n + bn;
      const bf16* g1 = g0 + N;
      const v8bf r0a = *(const v8bf*)g0, r0b = *(const v8bf*)(g0 + 8);
      const v8bf r1a = *(const v8bf*)g1, r1b = *(const v8bf*)(g1 + 8);
#pragma unroll
      for (int j = 0; j < 8; ++j) {
        v2bf pa; pa[0] = r0a[j]; pa[1] = r1a[j];
        *(v2bf*)(sBt + (size_t)(bn + j) * 72 + bk) = pa;
        v2bf pb; pb[0] = r0b[j]; pb[1] = r1b[j];
        *(v2bf*)(sBt + (size_t)(bn + 8 + j) * 72 + bk) = pb;
      }
    }
    if (k0 + 64 < K) {  // gfx1250 global_prefetch_b8 for next K-slab
      __builtin_prefetch(A + (size_t)(tile_m + ar) * K + (k0 + 64) + ac, 0, 1);
      __builtin_prefetch(Bm + (size_t)(k0 + 64 + bk) * N + tile_n + bn, 0, 1);
    }
    __syncthreads();
#pragma unroll
    for (int kk = 0; kk < 2; ++kk) {
      v16bf af[2], bfr[4];
#pragma unroll
      for (int i = 0; i < 2; ++i)
        af[i] = load_frag_a(sA, 64, wm * 32 + i * 16, kk * 32);
#pragma unroll
      for (int j = 0; j < 4; ++j)
        bfr[j] = load_frag_a(sBt, 72, wn * 64 + j * 16, kk * 32);
#pragma unroll
      for (int i = 0; i < 2; ++i)
#pragma unroll
        for (int j = 0; j < 4; ++j) acc[i][j] = wmma_bf16(af[i], bfr[j], acc[i][j]);
    }
    __syncthreads();
  }
#pragma unroll
  for (int i = 0; i < 2; ++i)
#pragma unroll
    for (int j = 0; j < 4; ++j)
#pragma unroll
      for (int r = 0; r < 8; ++r) {
        int mm = tile_m + wm * 32 + i * 16 + r + hi * 8;
        int nn = tile_n + wn * 64 + j * 16 + cc;
        Cm[(size_t)mm * N + nn] = acc[i][j][r];
      }
}

// ---------------- rotary + scale + (B,T,H,DK)->(B,H,T,DK) bf16 ----------------
__global__ __launch_bounds__(64) void rotary_transpose(const float* __restrict__ in,
                                                       bf16* __restrict__ out,
                                                       float scale) {
  const int h = blockIdx.x % H_;
  const int t = (blockIdx.x / H_) % T_;
  const int b = blockIdx.x / (H_ * T_);
  const int i = threadIdx.x;  // half index 0..63
  const float* src = in + (((size_t)b * T_ + t) * H_ + h) * DK_;
  const float x1 = src[i], x2 = src[i + 64];
  const float inv = __powf(10000.f, -(float)i * (1.f / 64.f));
  float sn, csn;
  __sincosf((float)t * inv, &sn, &csn);
  bf16* dst = out + (((size_t)b * H_ + h) * T_ + t) * DK_;
  dst[i]      = (bf16)((x1 * csn - x2 * sn) * scale);
  dst[i + 64] = (bf16)((x2 * csn + x1 * sn) * scale);
}

// ---------------- (B,T,H,DV) f32 -> (B,H,T,DV) bf16 ----------------
__global__ __launch_bounds__(256) void transpose_v(const float* __restrict__ in,
                                                   bf16* __restrict__ out) {
  const int h = blockIdx.x % H_;
  const int t = (blockIdx.x / H_) % T_;
  const int b = blockIdx.x / (H_ * T_);
  const int d = threadIdx.x;
  out[(((size_t)b * H_ + h) * T_ + t) * DV_ + d] =
      (bf16)in[(((size_t)b * T_ + t) * H_ + h) * DV_ + d];
}

// ---------------- per-chunk retention: scores, inner, local state U ----------------
// Block = one (b,h,chunk). 8 waves. All three matmuls on WMMA.
// Q/K chunk loads use gfx1250 GLOBAL_LOAD_ASYNC_TO_LDS_B128 when available.
__global__ __launch_bounds__(256) void retention_chunk(
    const bf16* __restrict__ qcb, const bf16* __restrict__ kcb,
    const bf16* __restrict__ vcb, float* __restrict__ attn,
    float* __restrict__ U) {
  __shared__ __align__(16) bf16 sQ[C_ * DK_];          // 16 KB
  __shared__ __align__(16) bf16 sK[C_ * DK_];          // 16 KB
  __shared__ __align__(16) bf16 sKt[DK_ * (C_ + 8)];   // 18 KB  (K^T * k_decay)
  __shared__ __align__(16) bf16 sVt[DV_ * (C_ + 8)];   // 36 KB  (V^T)
  __shared__ __align__(16) bf16 sS[C_ * (C_ + 8)];     // 9 KB   (masked scores)
  const int tid = threadIdx.x;
  const int wave = tid >> 5, wm = wave >> 1, wn = wave & 1;
  const int lane = tid & 31, hi = lane >> 4, cc = lane & 15;
  const int n = blockIdx.x % NCH_;
  const int h = (blockIdx.x / NCH_) % H_;
  const int b = blockIdx.x / (NCH_ * H_);
  const float gam = 1.f - exp2f(-5.f - (float)h);
  const float lg = __logf(gam);
  const size_t qbase = (((size_t)b * H_ + h) * T_ + (size_t)n * C_) * DK_;
  const size_t vbase = (((size_t)b * H_ + h) * T_ + (size_t)n * C_) * DV_;
#if __has_builtin(__builtin_amdgcn_global_load_async_to_lds_b128)
  {  // async DMA: 64x128 bf16 contiguous chunk -> LDS, no VGPR round-trip
    auto gq = (__attribute__((address_space(1))) i32x4*)(void*)(qcb + qbase);
    auto gk = (__attribute__((address_space(1))) i32x4*)(void*)(kcb + qbase);
    auto lq = (__attribute__((address_space(3))) i32x4*)sQ;
    auto lk = (__attribute__((address_space(3))) i32x4*)sK;
#pragma unroll
    for (int i = 0; i < 4; ++i) {
      const int o = tid * 4 + i;  // 16B units
      __builtin_amdgcn_global_load_async_to_lds_b128(gq + o, lq + o, 0, 0);
      __builtin_amdgcn_global_load_async_to_lds_b128(gk + o, lk + o, 0, 0);
    }
  }
#else
  {  // fallback: contiguous 64x128 chunks via bulk b128 copies
    const uint4* gq = (const uint4*)(qcb + qbase);
    const uint4* gk = (const uint4*)(kcb + qbase);
    uint4* lq = (uint4*)sQ;
    uint4* lk = (uint4*)sK;
#pragma unroll
    for (int i = 0; i < 4; ++i) {
      lq[tid * 4 + i] = gq[tid * 4 + i];
      lk[tid * 4 + i] = gk[tid * 4 + i];
    }
  }
#endif
  for (int idx = tid; idx < C_ * DK_; idx += 256) {  // K^T scaled by k_decay
    int c = idx >> 7, d = idx & 127;
    float kd = __expf(lg * (float)(C_ - 1 - c));
    sKt[d * (C_ + 8) + c] = (bf16)((float)kcb[qbase + idx] * kd);
  }
  for (int idx = tid; idx < C_ * DV_; idx += 256) {  // V^T
    int c = idx >> 8, d = idx & 255;
    sVt[d * (C_ + 8) + c] = vcb[vbase + idx];
  }
#if __has_builtin(__builtin_amdgcn_global_load_async_to_lds_b128)
#if __has_builtin(__builtin_amdgcn_s_wait_asynccnt)
  __builtin_amdgcn_s_wait_asynccnt(0);
#else
  asm volatile("s_wait_asynccnt 0x0" ::: "memory");
#endif
#endif
  __syncthreads();
  // -------- phase 1: scores = Q @ K^T (64x64), decay mask, -> sS bf16 --------
  v8f sc[2] = {};
#pragma unroll
  for (int k0 = 0; k0 < DK_; k0 += 32) {
    v16bf a = load_frag_a(sQ, DK_, wm * 16, k0);
#pragma unroll
    for (int j = 0; j < 2; ++j) {
      v16bf bb = load_frag_a(sK, DK_, (wn * 2 + j) * 16, k0);  // B=K^T: contiguous
      sc[j] = wmma_bf16(a, bb, sc[j]);
    }
  }
#pragma unroll
  for (int j = 0; j < 2; ++j)
#pragma unroll
    for (int r = 0; r < 8; ++r) {
      int i = wm * 16 + r + hi * 8;
      int jj = (wn * 2 + j) * 16 + cc;
      float msk = (i >= jj) ? __expf(lg * (float)(i - jj)) : 0.f;
      sS[i * (C_ + 8) + jj] = (bf16)(sc[j][r] * msk);
    }
  __syncthreads();
  // -------- phase 2: inner = scores @ V (64x256) -> attn (B,T,H,DV) --------
  {
    v8f o8[8] = {};
#pragma unroll
    for (int k0 = 0; k0 < C_; k0 += 32) {
      v16bf a = load_frag_a(sS, C_ + 8, wm * 16, k0);
#pragma unroll
      for (int j = 0; j < 8; ++j) {
        v16bf bb = load_frag_a(sVt, C_ + 8, (wn * 8 + j) * 16, k0);
        o8[j] = wmma_bf16(a, bb, o8[j]);
      }
    }
#pragma unroll
    for (int j = 0; j < 8; ++j)
#pragma unroll
      for (int r = 0; r < 8; ++r) {
        int i = wm * 16 + r + hi * 8;
        int dv = (wn * 8 + j) * 16 + cc;
        attn[(((size_t)b * T_ + (size_t)n * C_ + i) * H_ + h) * DV_ + dv] = o8[j][r];
      }
  }
  // -------- phase 3: U = (K*decay)^T @ V (128x256) -> state workspace --------
  {
    v8f u8[2][8] = {};
#pragma unroll
    for (int k0 = 0; k0 < C_; k0 += 32) {
      v16bf a0 = load_frag_a(sKt, C_ + 8, (wm * 2 + 0) * 16, k0);
      v16bf a1 = load_frag_a(sKt, C_ + 8, (wm * 2 + 1) * 16, k0);
#pragma unroll
      for (int j = 0; j < 8; ++j) {
        v16bf bb = load_frag_a(sVt, C_ + 8, (wn * 8 + j) * 16, k0);
        u8[0][j] = wmma_bf16(a0, bb, u8[0][j]);
        u8[1][j] = wmma_bf16(a1, bb, u8[1][j]);
      }
    }
    float* ub = U + (((size_t)b * H_ + h) * NCH_ + n) * (size_t)(DK_ * DV_);
#pragma unroll
    for (int mi = 0; mi < 2; ++mi)
#pragma unroll
      for (int j = 0; j < 8; ++j)
#pragma unroll
        for (int r = 0; r < 8; ++r) {
          int m = (wm * 2 + mi) * 16 + r + hi * 8;
          int nn = (wn * 8 + j) * 16 + cc;
          ub[(size_t)m * DV_ + nn] = u8[mi][j][r];
        }
  }
}

// ---------------- exclusive prefix-scan of chunk states (in place) ----------------
// One block per (b,h). State (128x256 f32) lives in 128 VGPRs per thread.
__global__ __launch_bounds__(256) void retention_scan(float* __restrict__ U) {
  const int tid = threadIdx.x;
  const int h = blockIdx.x % H_;
  const float gam = 1.f - exp2f(-5.f - (float)h);
  const float cross = __powf(gam, (float)C_);
  float* base = U + (size_t)blockIdx.x * NCH_ * (size_t)(DK_ * DV_);
  float S[128];
#pragma unroll
  for (int r = 0; r < 128; ++r) S[r] = 0.f;
  for (int n = 0; n < NCH_; ++n) {
    float* u = base + (size_t)n * (DK_ * DV_);
#pragma unroll
    for (int r = 0; r < 128; ++r) {
      const int i = tid + (r << 8);
      const float un = u[i];
      u[i] = S[r];                 // exclusive prefix -> S_prev for chunk n
      S[r] = cross * S[r] + un;    // include chunk n for the next one
    }
  }
}

// ---------------- inter = (Q*q_decay) @ S_prev, accumulate into attn ----------------
__global__ __launch_bounds__(256) void retention_inter(
    const bf16* __restrict__ qcb, const float* __restrict__ Sprev,
    float* __restrict__ attn) {
  __shared__ __align__(16) bf16 sQd[C_ * DK_];         // 16 KB (q * q_decay)
  __shared__ __align__(16) bf16 sSt[DV_ * (DK_ + 8)];  // 68 KB (S^T)
  const int tid = threadIdx.x;
  const int wave = tid >> 5, wm = wave >> 1, wn = wave & 1;
  const int lane = tid & 31, hi = lane >> 4, cc = lane & 15;
  const int n = blockIdx.x % NCH_;
  const int h = (blockIdx.x / NCH_) % H_;
  const int b = blockIdx.x / (NCH_ * H_);
  const float gam = 1.f - exp2f(-5.f - (float)h);
  const float lg = __logf(gam);
  const size_t qbase = (((size_t)b * H_ + h) * T_ + (size_t)n * C_) * DK_;
  const float* sp = Sprev + (((size_t)b * H_ + h) * NCH_ + n) * (size_t)(DK_ * DV_);
  for (int idx = tid; idx < C_ * DK_; idx += 256) {
    int r = idx >> 7;
    float qd = __expf(lg * (float)(r + 1));
    sQd[idx] = (bf16)((float)qcb[qbase + idx] * qd);
  }
  for (int idx = tid; idx < DK_ * DV_; idx += 256) {
    int k = idx >> 8, nn = idx & 255;
    sSt[nn * (DK_ + 8) + k] = (bf16)sp[idx];
  }
  __syncthreads();
  v8f o8[8] = {};
#pragma unroll
  for (int k0 = 0; k0 < DK_; k0 += 32) {
    v16bf a = load_frag_a(sQd, DK_, wm * 16, k0);
#pragma unroll
    for (int j = 0; j < 8; ++j) {
      v16bf bb = load_frag_a(sSt, DK_ + 8, (wn * 8 + j) * 16, k0);
      o8[j] = wmma_bf16(a, bb, o8[j]);
    }
  }
#pragma unroll
  for (int j = 0; j < 8; ++j)
#pragma unroll
    for (int r = 0; r < 8; ++r) {
      int i = wm * 16 + r + hi * 8;
      int dv = (wn * 8 + j) * 16 + cc;
      size_t idx = (((size_t)b * T_ + (size_t)n * C_ + i) * H_ + h) * DV_ + dv;
      attn[idx] += o8[j][r];
    }
}

// ---------------- gated RMSNorm + SiLU: o2 = rms(o)*w * g*sigmoid(g) ----------------
__global__ __launch_bounds__(256) void gate_rmsnorm(
    const float* __restrict__ attn, const float* __restrict__ gf,
    const float* __restrict__ rmsw, bf16* __restrict__ o2) {
  __shared__ float red[256];
  const int tid = threadIdx.x;
  const size_t row = blockIdx.x;  // (b*T + t)*H + h ; flat layouts coincide
  const float o = attn[row * DV_ + tid];
  red[tid] = o * o;
  __syncthreads();
  for (int s = 128; s > 0; s >>= 1) {
    if (tid < s) red[tid] += red[tid + s];
    __syncthreads();
  }
  const float rr = __frsqrt_rn(red[0] * (1.f / DV_) + 1e-5f);
  const float g = gf[row * DV_ + tid];
  const float sig = 1.f / (1.f + __expf(-g));
  o2[row * DV_ + tid] = (bf16)(o * rr * rmsw[tid] * g * sig);
}

// ---------------- y = x + proj; LayerNorm over D ----------------
__global__ __launch_bounds__(256) void residual_layernorm(
    const float* __restrict__ x, const float* __restrict__ proj,
    const float* __restrict__ w, const float* __restrict__ bia,
    float* __restrict__ out) {
  __shared__ float rsum[256];
  __shared__ float rsq[256];
  const int tid = threadIdx.x;
  const size_t row = blockIdx.x;
  float y[4], s = 0.f, sq = 0.f;
#pragma unroll
  for (int j = 0; j < 4; ++j) {
    int d = tid + j * 256;
    float v = x[row * D_ + d] + proj[row * D_ + d];
    y[j] = v; s += v; sq += v * v;
  }
  rsum[tid] = s; rsq[tid] = sq;
  __syncthreads();
  for (int st = 128; st > 0; st >>= 1) {
    if (tid < st) { rsum[tid] += rsum[tid + st]; rsq[tid] += rsq[tid + st]; }
    __syncthreads();
  }
  const float mu = rsum[0] * (1.f / D_);
  const float var = rsq[0] * (1.f / D_) - mu * mu;
  const float inv = __frsqrt_rn(var + 1e-5f);
#pragma unroll
  for (int j = 0; j < 4; ++j) {
    int d = tid + j * 256;
    out[row * D_ + d] = (y[j] - mu) * inv * w[d] + bia[d];
  }
}

extern "C" void kernel_launch(void* const* d_in, const int* in_sizes, int n_in,
                              void* d_out, int out_size, void* d_ws, size_t ws_size,
                              hipStream_t stream) {
  (void)in_sizes; (void)n_in; (void)out_size; (void)ws_size;
  const float* x    = (const float*)d_in[0];
  const float* Wq   = (const float*)d_in[1];
  const float* Wk   = (const float*)d_in[2];
  const float* Wv   = (const float*)d_in[3];
  const float* Wg   = (const float*)d_in[4];
  const float* Wo   = (const float*)d_in[5];
  const float* rmsw = (const float*)d_in[6];
  const float* lnw  = (const float*)d_in[7];
  const float* lnb  = (const float*)d_in[8];
  float* out = (float*)d_out;

  const size_t BT = (size_t)B_ * T_;
  char* ws = (char*)d_ws;
  size_t off = 0;
  auto take = [&](size_t bytes) -> char* {
    char* p = ws + off;
    off += (bytes + 255) & ~(size_t)255;
    return p;
  };
  bf16* xb   = (bf16*)take(BT * D_ * 2);
  bf16* wqb  = (bf16*)take((size_t)D_ * D_ * 2);
  bf16* wkb  = (bf16*)take((size_t)D_ * D_ * 2);
  bf16* wvb  = (bf16*)take((size_t)D_ * 2 * D_ * 2);
  bf16* wgb  = (bf16*)take((size_t)D_ * 2 * D_ * 2);
  bf16* wob  = (bf16*)take((size_t)2 * D_ * D_ * 2);
  float* qf  = (float*)take(BT * D_ * 4);
  float* kf  = (float*)take(BT * D_ * 4);
  float* vf  = (float*)take(BT * 2 * D_ * 4);
  float* gf  = (float*)take(BT * 2 * D_ * 4);
  bf16* qcb  = (bf16*)take(BT * D_ * 2);
  bf16* kcb  = (bf16*)take(BT * D_ * 2);
  bf16* vcb  = (bf16*)take(BT * 2 * D_ * 2);
  float* U   = (float*)take((size_t)B_ * H_ * NCH_ * DK_ * DV_ * 4);
  // Lifetime-based aliases (stream-ordered, no overlap of live ranges):
  float* attn = vf;         // vf dead after transpose_v
  bf16*  o2b  = (bf16*)qf;  // qf dead after rotary_transpose(q)
  float* yf   = kf;         // kf dead after rotary_transpose(k)

  auto cvt = [&](const float* src, bf16* dst, size_t n) {
    int n4 = (int)(n / 4);
    cvt_f32_to_bf16<<<(n4 + 255) / 256, 256, 0, stream>>>(src, dst, n4);
  };
  cvt(x,  xb,  BT * D_);
  cvt(Wq, wqb, (size_t)D_ * D_);
  cvt(Wk, wkb, (size_t)D_ * D_);
  cvt(Wv, wvb, (size_t)D_ * 2 * D_);
  cvt(Wg, wgb, (size_t)D_ * 2 * D_);
  cvt(Wo, wob, (size_t)2 * D_ * D_);

  const dim3 blk(256);
  gemm_bf16_wmma<<<dim3(D_ / 128, (int)(BT / 128)), blk, 0, stream>>>(xb, wqb, qf, (int)BT, D_, D_);
  gemm_bf16_wmma<<<dim3(D_ / 128, (int)(BT / 128)), blk, 0, stream>>>(xb, wkb, kf, (int)BT, D_, D_);
  gemm_bf16_wmma<<<dim3(2 * D_ / 128, (int)(BT / 128)), blk, 0, stream>>>(xb, wvb, vf, (int)BT, 2 * D_, D_);
  gemm_bf16_wmma<<<dim3(2 * D_ / 128, (int)(BT / 128)), blk, 0, stream>>>(xb, wgb, gf, (int)BT, 2 * D_, D_);

  const int bth = B_ * T_ * H_;
  rotary_transpose<<<bth, 64, 0, stream>>>(qf, qcb, 0.08838834764831843f);  // DK^-0.5
  rotary_transpose<<<bth, 64, 0, stream>>>(kf, kcb, 1.0f);
  transpose_v<<<bth, 256, 0, stream>>>(vf, vcb);

  retention_chunk<<<B_ * H_ * NCH_, 256, 0, stream>>>(qcb, kcb, vcb, attn, U);
  retention_scan<<<B_ * H_, 256, 0, stream>>>(U);
  retention_inter<<<B_ * H_ * NCH_, 256, 0, stream>>>(qcb, U, attn);

  gate_rmsnorm<<<bth, 256, 0, stream>>>(attn, gf, rmsw, o2b);
  gemm_bf16_wmma<<<dim3(D_ / 128, (int)(BT / 128)), blk, 0, stream>>>(o2b, wob, yf, (int)BT, D_, 2 * D_);
  residual_layernorm<<<(int)BT, 256, 0, stream>>>(x, yf, lnw, lnb, out);
}
